// Discriminator_30554397343836
// MI455X (gfx1250) — matrix operand order repaired and checked by reading, not compiled, not run on hardware
//
#include <hip/hip_runtime.h>
#include <hip/hip_bf16.h>

#define Bsz  64
#define Tlen 256
#define Lsnd 65536
#define Hdim 256

typedef __attribute__((ext_vector_type(16))) __bf16 v16bf;
typedef __attribute__((ext_vector_type(8)))  float  v8f;
typedef __attribute__((ext_vector_type(8)))  int    v8i;
typedef __attribute__((ext_vector_type(4)))  int    v4i;

__device__ __forceinline__ unsigned short f2bf(float x) {
    unsigned u = __float_as_uint(x);
    u += 0x7FFFu + ((u >> 16) & 1u);   // round-to-nearest-even
    return (unsigned short)(u >> 16);
}
__device__ __forceinline__ float sigmoidf_(float x) { return 1.0f / (1.0f + __expf(-x)); }

// ---------------------------------------------------------------------------
// Pack a [1024,256] f32 weight matrix into bf16 B-matrix WMMA fragments.
// B tile is 32(K) x 16(N); fragment for (nt, kb) stored per-lane contiguous:
//   dword j of lane l lives at ((nt*8+kb)*32 + l)*8 + j
// Layout (ISA 7.12.2, 16-bit): VGPR j<4: lanes0-15 hold K=2j,2j+1 (N=lane),
// lanes16-31 hold K=8+2j,8+2j+1; VGPR j>=4: lanes0-15 K=8+2j,.., lanes16-31 K=16+2j,..
// ---------------------------------------------------------------------------
__global__ __launch_bounds__(256) void pack_weights(const float* __restrict__ w,
                                                    unsigned short* __restrict__ dst) {
    int i = blockIdx.x * 256 + threadIdx.x;        // 0 .. 262143
    int n = i >> 8, k = i & 255;                   // W[n][k]
    int nt = n >> 4, nn = n & 15, kb = k >> 5, kk = k & 31;
    int hi = (kk >> 3) & 1, k2 = ((kk >> 4) << 3) | (kk & 7), j = k2 >> 1, half = kk & 1;
    int lane = nn + (hi ? 16 : 0);
    size_t idx = (((size_t)(nt * 8 + kb) * 32 + lane) * 8 + j) * 2 + half;
    dst[idx] = f2bf(w[i]);
}

// ---------------------------------------------------------------------------
// Frontend: gather fragment, conv1 (3 dots of len 2046), conv2 (32ch k3 pad1
// + relu), conv3 (128ch k3), concat interval, linear 129->256.  One block of
// 256 threads per (t,b).  Output written as bf16 A-matrix WMMA fragments:
// A tile 16(M) x 32(K); element (m,k): group=k>>4, r=k&15, hi=(r>=8),
// vgpr=4*group+((r&7)>>1), lane=m+16*hi, half=k&1.
// ---------------------------------------------------------------------------
__global__ __launch_bounds__(256) void frontend_kernel(
    const int* __restrict__ alpha, const float* __restrict__ sound,
    const float* __restrict__ c1w, const float* __restrict__ c1b,
    const float* __restrict__ c2w, const float* __restrict__ c2b,
    const float* __restrict__ c3w, const float* __restrict__ c3b,
    const float* __restrict__ linw, const float* __restrict__ linb,
    unsigned short* __restrict__ xfrag)
{
    __shared__ float red0[256], red1[256], red2[256];
    __shared__ float y1p[5];      // [0, y1[0], y1[1], y1[2], 0]
    __shared__ float y2sh[96];    // [32ch][3l]
    __shared__ float feat[129];   // [interval, y3[0..127]]
    const int bid = blockIdx.x;
    const int t = bid >> 6;
    const int b = bid & 63;
    const int tid = threadIdx.x;
    const int a0 = alpha[b * Tlen + t];
    const float* snd = sound + (size_t)b * Lsnd;

    float s0 = 0.f, s1 = 0.f, s2 = 0.f;
    for (int k = tid; k < 2046; k += 256) {
        float w = c1w[k];
        int base = a0 - 1024 + k;  // frag[k] = padded[a0+k] = sound[a0+k-1024]
        float f0 = (base     >= 0 && base     < Lsnd) ? snd[base]     : 0.f;
        float f1 = (base + 1 >= 0 && base + 1 < Lsnd) ? snd[base + 1] : 0.f;
        float f2 = (base + 2 >= 0 && base + 2 < Lsnd) ? snd[base + 2] : 0.f;
        s0 += w * f0; s1 += w * f1; s2 += w * f2;
    }
    red0[tid] = s0; red1[tid] = s1; red2[tid] = s2;
    __syncthreads();
    for (int st = 128; st > 0; st >>= 1) {
        if (tid < st) {
            red0[tid] += red0[tid + st];
            red1[tid] += red1[tid + st];
            red2[tid] += red2[tid + st];
        }
        __syncthreads();
    }
    if (tid == 0) {
        float bb = c1b[0];
        y1p[0] = 0.f; y1p[1] = red0[0] + bb; y1p[2] = red1[0] + bb;
        y1p[3] = red2[0] + bb; y1p[4] = 0.f;
    }
    __syncthreads();
    if (tid < 96) {               // conv2 + relu
        int cch = tid / 3, l = tid % 3;
        float a = c2b[cch];
        #pragma unroll
        for (int k = 0; k < 3; ++k) a += c2w[cch * 3 + k] * y1p[l + k];
        y2sh[cch * 3 + l] = fmaxf(a, 0.f);
    }
    if (tid == 128) {             // interval feature
        int ap = (t > 0) ? alpha[b * Tlen + t - 1] : 0;
        feat[0] = (float)(a0 - ap);
    }
    __syncthreads();
    if (tid < 128) {              // conv3 (no relu)
        float a = c3b[tid];
        for (int cch = 0; cch < 32; ++cch)
            #pragma unroll
            for (int k = 0; k < 3; ++k)
                a += y2sh[cch * 3 + k] * c3w[(tid * 32 + cch) * 3 + k];
        feat[1 + tid] = a;
    }
    __syncthreads();
    {                              // linear 129 -> 256, write swizzled bf16
        float a = linb[tid];
        const float* wr = linw + tid * 129;
        for (int j = 0; j < 129; ++j) a += feat[j] * wr[j];
        int f = tid;
        int kb = f >> 5, kk = f & 31;
        int mt = b >> 4, m = b & 15;
        int group = kk >> 4, rr = kk & 15;
        int lhi = (rr >= 8);
        int jj = (rr & 7) >> 1;
        int half = kk & 1;
        int vg = 4 * group + jj;
        int al = m + (lhi ? 16 : 0);
        size_t idx = ((((size_t)t * 4 + mt) * 8 + kb) * 32 + al) * 16 + vg * 2 + half;
        xfrag[idx] = f2bf(a);
    }
}

// ---------------------------------------------------------------------------
// LSTM: single persistent workgroup of 1024 threads (32 waves).
// Per step: g[64,1024] = x_t @ Wih^T + h @ Whh^T + b, via bf16 WMMA.
// Wave w owns N-tiles {2w, 2w+1}, one per pass (bounds register pressure).
// Per k-block: batch-load all 4 M-tile A fragments into distinct registers,
// then issue 4 independent back-to-back WMMAs so loads overlap the matrix
// pipe instead of serializing on s_wait_loadcnt 0.
// LDS: gates f32 [64][1024] (256 KB) + h bf16 A-fragments (32 KB).
// c kept in registers: thread tid owns (b=tid>>4, units (tid&15)*16..+15).
// ---------------------------------------------------------------------------
__global__ __launch_bounds__(1024) void lstm_kernel(
    const int* __restrict__ xfrag, const int* __restrict__ wih,
    const int* __restrict__ whh, const float* __restrict__ b_ih,
    const float* __restrict__ b_hh, float* __restrict__ hout)
{
    extern __shared__ char smem[];
    float* gates = (float*)smem;                                   // [64][1024]
    unsigned short* hswz = (unsigned short*)(smem + Bsz * 1024 * sizeof(float));
    int* hswz32 = (int*)hswz;

    const int tid = threadIdx.x;
    const int wave = tid >> 5, lane = tid & 31;
    const int nt0 = wave * 2;

    for (int i = tid; i < (Bsz * Hdim) / 2; i += 1024) hswz32[i] = 0;  // h0 = 0

    float c[16];
    #pragma unroll
    for (int i = 0; i < 16; ++i) c[i] = 0.f;
    const int cb = tid >> 4;
    const int cu0 = (tid & 15) << 4;

    float bias[2];
    int act_tanh[2];
    #pragma unroll
    for (int q = 0; q < 2; ++q) {
        int n = (nt0 + q) * 16 + (lane & 15);
        bias[q] = b_ih[n] + b_hh[n];
        act_tanh[q] = (((nt0 + q) >> 4) == 2);   // gates [512,768) use tanh
    }
    __syncthreads();

    #pragma unroll 1
    for (int t = 0; t < Tlen; ++t) {
        #pragma unroll 1
        for (int pass = 0; pass < 2; ++pass) {
            const int nt = nt0 + pass;
            v8f acc[4];
            #pragma unroll
            for (int mt = 0; mt < 4; ++mt)
                #pragma unroll
                for (int r = 0; r < 8; ++r) acc[mt][r] = 0.f;

            // ---- x_t @ Wih^T : B frag from L2, 4 A frags batched from global
            // mt stride in dwords: 8(kb) * 32(lanes) * 8(dwords) = 2048
            #pragma unroll 1
            for (int kb = 0; kb < 8; ++kb) {
                size_t wb = (((size_t)nt * 8 + kb) * 32 + lane) * 8;
                v16bf bw = __builtin_bit_cast(v16bf, *(const v8i*)(wih + wb));
                if (kb < 7) __builtin_prefetch((const void*)(wih + wb + 256), 0, 1);
                const int* xp = xfrag + (((size_t)t * 32 + kb) * 32 + lane) * 8;
                v16bf a0 = __builtin_bit_cast(v16bf, *(const v8i*)(xp));
                v16bf a1 = __builtin_bit_cast(v16bf, *(const v8i*)(xp + 2048));
                v16bf a2 = __builtin_bit_cast(v16bf, *(const v8i*)(xp + 4096));
                v16bf a3 = __builtin_bit_cast(v16bf, *(const v8i*)(xp + 6144));
                acc[0] = __builtin_amdgcn_wmma_f32_16x16x32_bf16(
                    false, a0, false, bw, (short)0, acc[0], false, false);
                acc[1] = __builtin_amdgcn_wmma_f32_16x16x32_bf16(
                    false, a1, false, bw, (short)0, acc[1], false, false);
                acc[2] = __builtin_amdgcn_wmma_f32_16x16x32_bf16(
                    false, a2, false, bw, (short)0, acc[2], false, false);
                acc[3] = __builtin_amdgcn_wmma_f32_16x16x32_bf16(
                    false, a3, false, bw, (short)0, acc[3], false, false);
            }
            // ---- h @ Whh^T : B frag from L2, 4 A frags batched from LDS
            #pragma unroll 1
            for (int kb = 0; kb < 8; ++kb) {
                size_t wb = (((size_t)nt * 8 + kb) * 32 + lane) * 8;
                v16bf bw = __builtin_bit_cast(v16bf, *(const v8i*)(whh + wb));
                if (kb < 7) __builtin_prefetch((const void*)(whh + wb + 256), 0, 1);
                const int* hp = hswz32 + (kb * 32 + lane) * 8;
                v16bf a0, a1, a2, a3;
                {
                    v4i l0 = *(const v4i*)(hp),        l1 = *(const v4i*)(hp + 4);
                    v4i l2 = *(const v4i*)(hp + 2048), l3 = *(const v4i*)(hp + 2052);
                    v4i l4 = *(const v4i*)(hp + 4096), l5 = *(const v4i*)(hp + 4100);
                    v4i l6 = *(const v4i*)(hp + 6144), l7 = *(const v4i*)(hp + 6148);
                    a0 = __builtin_bit_cast(v16bf, (v8i){l0[0],l0[1],l0[2],l0[3],l1[0],l1[1],l1[2],l1[3]});
                    a1 = __builtin_bit_cast(v16bf, (v8i){l2[0],l2[1],l2[2],l2[3],l3[0],l3[1],l3[2],l3[3]});
                    a2 = __builtin_bit_cast(v16bf, (v8i){l4[0],l4[1],l4[2],l4[3],l5[0],l5[1],l5[2],l5[3]});
                    a3 = __builtin_bit_cast(v16bf, (v8i){l6[0],l6[1],l6[2],l6[3],l7[0],l7[1],l7[2],l7[3]});
                }
                acc[0] = __builtin_amdgcn_wmma_f32_16x16x32_bf16(
                    false, a0, false, bw, (short)0, acc[0], false, false);
                acc[1] = __builtin_amdgcn_wmma_f32_16x16x32_bf16(
                    false, a1, false, bw, (short)0, acc[1], false, false);
                acc[2] = __builtin_amdgcn_wmma_f32_16x16x32_bf16(
                    false, a2, false, bw, (short)0, acc[2], false, false);
                acc[3] = __builtin_amdgcn_wmma_f32_16x16x32_bf16(
                    false, a3, false, bw, (short)0, acc[3], false, false);
            }
            // ---- bias + activation, scatter to gates LDS
            // (C/D layout: VGPR r -> M = r + 8*(lane>>4), N = lane&15)
            const int hi = lane >> 4;
            const int gcol = nt * 16 + (lane & 15);
            #pragma unroll
            for (int mt = 0; mt < 4; ++mt) {
                #pragma unroll
                for (int r = 0; r < 8; ++r) {
                    float v = acc[mt][r] + bias[pass];
                    v = act_tanh[pass] ? tanhf(v) : sigmoidf_(v);
                    gates[(16 * mt + r + 8 * hi) * 1024 + gcol] = v;
                }
            }
        }
        __syncthreads();

        // c/h update; re-swizzle h into bf16 A-fragment LDS buffer
        #pragma unroll
        for (int i = 0; i < 16; ++i) {
            int u = cu0 + i;
            float iv = gates[cb * 1024 + u];
            float fv = gates[cb * 1024 + 256 + u];
            float gv = gates[cb * 1024 + 512 + u];
            float ov = gates[cb * 1024 + 768 + u];
            c[i] = fv * c[i] + iv * gv;
            float h = ov * tanhf(c[i]);
            int mt = cb >> 4, m = cb & 15, kb = u >> 5, kk = u & 31;
            int group = kk >> 4, rr = kk & 15;
            int lhi = (rr >= 8);
            int jj = (rr & 7) >> 1;
            int half = kk & 1;
            int vg = 4 * group + jj;
            int al = m + (lhi ? 16 : 0);
            hswz[(((mt * 8 + kb) * 32 + al) * 8 + vg) * 2 + half] = f2bf(h);
            if (t == Tlen - 1) hout[cb * Hdim + u] = h;
        }
        __syncthreads();
    }
}

// ---------------------------------------------------------------------------
// Head: sigmoid(relu(h @ out1^T + b1) @ out2^T + b2) -> [64] floats
// ---------------------------------------------------------------------------
__global__ __launch_bounds__(64) void head_kernel(
    const float* __restrict__ hout, const float* __restrict__ o1w,
    const float* __restrict__ o1b, const float* __restrict__ o2w,
    const float* __restrict__ o2b, float* __restrict__ out)
{
    int b = threadIdx.x;
    const float* h = hout + b * Hdim;
    float acc2 = o2b[0];
    for (int j = 0; j < 32; ++j) {
        float a = o1b[j];
        const float* wr = o1w + j * Hdim;
        for (int k = 0; k < Hdim; ++k) a += wr[k] * h[k];
        acc2 += fmaxf(a, 0.f) * o2w[j];
    }
    out[b] = 1.0f / (1.0f + __expf(-acc2));
}

extern "C" void kernel_launch(void* const* d_in, const int* in_sizes, int n_in,
                              void* d_out, int out_size, void* d_ws, size_t ws_size,
                              hipStream_t stream) {
    (void)in_sizes; (void)n_in; (void)out_size; (void)ws_size;
    const int*   alpha = (const int*)  d_in[0];
    const float* sound = (const float*)d_in[1];
    const float* c1w   = (const float*)d_in[2];
    const float* c1b   = (const float*)d_in[3];
    const float* c2w   = (const float*)d_in[4];
    const float* c2b   = (const float*)d_in[5];
    const float* c3w   = (const float*)d_in[6];
    const float* c3b   = (const float*)d_in[7];
    const float* linw  = (const float*)d_in[8];
    const float* linb  = (const float*)d_in[9];
    const float* w_ih  = (const float*)d_in[10];
    const float* w_hh  = (const float*)d_in[11];
    const float* b_ih  = (const float*)d_in[12];
    const float* b_hh  = (const float*)d_in[13];
    const float* o1w   = (const float*)d_in[14];
    const float* o1b   = (const float*)d_in[15];
    const float* o2w   = (const float*)d_in[16];
    const float* o2b   = (const float*)d_in[17];
    float* out = (float*)d_out;

    char* ws = (char*)d_ws;
    unsigned short* xfrag = (unsigned short*)ws;                            // 8 MiB: T*B*256 bf16 A-frags
    unsigned short* wihp  = (unsigned short*)(ws + (8u << 20));             // 512 KiB
    unsigned short* whhp  = (unsigned short*)(ws + (8u << 20) + (512u << 10));
    float*          hout  = (float*)(ws + (9u << 20));                      // 64 KiB

    pack_weights<<<1024, 256, 0, stream>>>(w_ih, wihp);
    pack_weights<<<1024, 256, 0, stream>>>(w_hh, whhp);
    frontend_kernel<<<Tlen * Bsz, 256, 0, stream>>>(alpha, sound, c1w, c1b, c2w, c2b,
                                                    c3w, c3b, linw, linb, xfrag);

    size_t lds_bytes = (size_t)Bsz * 1024 * sizeof(float) + (size_t)Bsz * Hdim * sizeof(unsigned short);
    (void)hipFuncSetAttribute((const void*)lstm_kernel,
                              hipFuncAttributeMaxDynamicSharedMemorySize, (int)lds_bytes);
    lstm_kernel<<<1, 1024, lds_bytes, stream>>>((const int*)xfrag, (const int*)wihp,
                                                (const int*)whhp, b_ih, b_hh, hout);
    head_kernel<<<1, 64, 0, stream>>>(hout, o1w, o1b, o2w, o2b, out);
}
